// SegHeader_59141699666149
// MI455X (gfx1250) — matrix operand order repaired and checked by reading, not compiled, not run on hardware
//
#include <hip/hip_runtime.h>
#include <hip/hip_bf16.h>

// ---------------------------------------------------------------------------
// Problem constants (from the reference)
// ---------------------------------------------------------------------------
static constexpr int N5 = 12000, N6 = 45000, N7 = 160000, N8 = 320000, Q = 200000;
static constexpr int C5 = 256, C6 = 128, C7 = 64;
static constexpr int F = 168, FP = 192, KT = 27, OUTC = 21, OUTP = 32;
static constexpr int NCH = 512;   // row-chunks for deterministic BN reduction

typedef __attribute__((ext_vector_type(16))) __bf16 bf16x16;
typedef __attribute__((ext_vector_type(8)))  float  f32x8;
typedef __attribute__((ext_vector_type(4)))  int    i32x4;
using bf16 = __bf16;

union ABFrag { uint4 u[2]; bf16x16 v; };

__device__ __forceinline__ uint4 ld128(const bf16* p) {
  return *reinterpret_cast<const uint4*>(p);
}

// ---------------------------------------------------------------------------
// Optional gfx1250 async global->LDS staging (fallback: through-register copy)
// ---------------------------------------------------------------------------
#if defined(__HIP_DEVICE_COMPILE__) && defined(__has_builtin)
# if __has_builtin(__builtin_amdgcn_global_load_async_to_lds_b128) && \
     __has_builtin(__builtin_amdgcn_s_wait_asynccnt)
#  define GCONV_ASYNC_LDS 1
# endif
#endif
#ifndef GCONV_ASYNC_LDS
# define GCONV_ASYNC_LDS 0
#endif

#if GCONV_ASYNC_LDS
#define AS1 __attribute__((address_space(1)))
#define AS3 __attribute__((address_space(3)))
#endif

// Copy one swizzled weight tap (elems bf16, elems/8 16B-chunks, multiple of 256)
// from global to LDS, block-cooperatively. No wait here; call stage_wait().
__device__ __forceinline__ void stage_tap(bf16* dst, const bf16* src, int elems, int tid) {
#if GCONV_ASYNC_LDS
  unsigned long long gaddr = (unsigned long long)src;
  unsigned laddr = (unsigned)(unsigned long long)dst;   // low 32 bits = LDS byte offset
  const int n = elems >> 3;
  for (int i = tid; i < n; i += 256) {
    __builtin_amdgcn_global_load_async_to_lds_b128(
        (AS1 i32x4*)(gaddr + (unsigned long long)i * 16u),
        (AS3 i32x4*)(laddr + (unsigned)i * 16u), 0, 0);
  }
#else
  const uint4* s = (const uint4*)src;
  uint4* d = (uint4*)dst;
  const int n = elems >> 3;
  for (int i = tid; i < n; i += 256) d[i] = s[i];
#endif
}

__device__ __forceinline__ void stage_wait() {
#if GCONV_ASYNC_LDS
  __builtin_amdgcn_s_wait_asynccnt(0);
#endif
}

// ---------------------------------------------------------------------------
// Core WMMA gather-GEMM:  out[n, :] = sum_k  x[ neigh[n,k], : ] @ W[k]
// x:   bf16 row-major [N, ck*32]
// WTS: bf16, fragment-swizzled: [ntaps][ck][NT][32 lanes][16]  (B fragments)
// out: f32 row-major [N, NT*16].  neigh==nullptr -> identity (dense GEMM).
// Block = 8 waves; wave = one 16-row tile x all NT*16 outputs. Weight tap is
// staged in LDS (double-buffered) and shared by all 8 waves.
// ---------------------------------------------------------------------------
template<int NT>
__global__ __launch_bounds__(256)
void k_gconv(const bf16* __restrict__ x, const int* __restrict__ neigh,
             int ntaps, const bf16* __restrict__ WTS,
             float* __restrict__ out, int N, int ck) {
  extern __shared__ char smem[];
  const int tapElems = ck * NT * 512;          // bf16 elements per tap
  bf16* sw0 = (bf16*)smem;
  bf16* sw1 = sw0 + tapElems;

  const int tid  = threadIdx.x;
  const int lane = tid & 31;
  const int wave = tid >> 5;
  const int row0 = (blockIdx.x * 8 + wave) * 16;   // may be >= N: compute dummy, skip stores
  const int l15  = lane & 15;
  const int half = lane >> 4;                  // 0 | 1
  const int ldx  = ck * 32;
  int nrow = row0 + l15; if (nrow > N - 1) nrow = N - 1;   // clamp (keeps EXEC uniform)

  const f32x8 zero = {0.f, 0.f, 0.f, 0.f, 0.f, 0.f, 0.f, 0.f};
  f32x8 acc[NT];
#pragma unroll
  for (int t = 0; t < NT; ++t) acc[t] = zero;

  stage_tap(sw0, WTS, tapElems, tid);
  stage_wait();
  __syncthreads();

  int g = neigh ? neigh[(size_t)nrow * ntaps] : nrow;
  for (int k = 0; k < ntaps; ++k) {
    const bf16* cur = (k & 1) ? sw1 : sw0;
    int gnext = g;
    if (k + 1 < ntaps) {                       // wave-uniform branch
      stage_tap((k & 1) ? sw0 : sw1, WTS + (size_t)(k + 1) * tapElems, tapElems, tid);
      gnext = neigh[(size_t)nrow * ntaps + (k + 1)];
      __builtin_prefetch(x + (size_t)gnext * ldx, 0, 0);   // prefetch next gathered row
    }
    const bf16* xr = x + (size_t)g * ldx;
    for (int kc = 0; kc < ck; ++kc) {
      // A fragment (16x32 bf16): lane<16 -> K 0..7 & 16..23; lane>=16 -> K 8..15 & 24..31
      ABFrag a;
      const bf16* ap = xr + kc * 32 + half * 8;
      a.u[0] = ld128(ap);
      a.u[1] = ld128(ap + 16);
      // B fragments: contiguous 32B per lane from LDS (pre-swizzled)
      const bf16* bbase = cur + (size_t)(kc * NT) * 512 + lane * 16;
#pragma unroll
      for (int t = 0; t < NT; ++t) {
        ABFrag b;
        const uint4* bp = (const uint4*)(bbase + t * 512);
        b.u[0] = bp[0];
        b.u[1] = bp[1];
        acc[t] = __builtin_amdgcn_wmma_f32_16x16x32_bf16(
            false, a.v, false, b.v, (short)0, acc[t], false, false);
      }
    }
    if (k + 1 < ntaps) { stage_wait(); __syncthreads(); }
    g = gnext;
  }

  // C/D layout: VGPR v -> M = v (lanes 0-15) | v+8 (lanes 16-31); N = l15
  const int ldo = NT * 16;
#pragma unroll
  for (int t = 0; t < NT; ++t) {
#pragma unroll
    for (int v = 0; v < 8; ++v) {
      const int m = row0 + half * 8 + v;
      if (m < N) out[(size_t)m * ldo + t * 16 + l15] = acc[t][v];
    }
  }
}

// ---------------------------------------------------------------------------
// Prep kernels
// ---------------------------------------------------------------------------
__global__ void k_cvt_pad(const float* __restrict__ in, bf16* __restrict__ out,
                          long long total, int C, int Cpad) {
  long long i = (long long)blockIdx.x * blockDim.x + threadIdx.x;
  if (i >= total) return;
  const int c = (int)(i % Cpad);
  const long long n = i / Cpad;
  out[i] = (c < C) ? (bf16)in[n * C + c] : (bf16)0.0f;
}

// W[t][ci][co] (f32) -> WTS fragment-swizzled [t][kc][nt][lane][e] (bf16, zero pad)
// fragment(t,kc,nt,lane): col = nt*16 + (lane&15); rows = kc*32 + (lane>=16?8:0) + (e&7) + (e>=8?16:0)
__global__ void k_repack_sw(const float* __restrict__ W, bf16* __restrict__ WTS,
                            long long total, int Cin, int Cout, int ck, int nt) {
  long long i = (long long)blockIdx.x * blockDim.x + threadIdx.x;
  if (i >= total) return;
  const int e = (int)(i & 15);
  long long r = i >> 4;
  const int ln = (int)(r & 31); r >>= 5;
  const int t  = (int)(r % nt); r /= nt;
  const int kc = (int)(r % ck);
  const int k  = (int)(r / ck);
  const int ci = kc * 32 + ((ln >> 4) ? 8 : 0) + (e & 7) + ((e >> 3) ? 16 : 0);
  const int co = t * 16 + (ln & 15);
  float v = (ci < Cin && co < Cout) ? W[((size_t)k * Cin + ci) * Cout + co] : 0.f;
  WTS[i] = (bf16)v;
}

// ---------------------------------------------------------------------------
// Deterministic BatchNorm statistics (two-stage, no atomics)
// x: f32 [N,192]; part: [NCH][2][192]; ss: [2][192] (scale, shift)
// ---------------------------------------------------------------------------
__global__ __launch_bounds__(192)
void k_bn_part(const float* __restrict__ x, int N, float* __restrict__ part) {
  const int c = threadIdx.x;
  const int chunk = blockIdx.x;
  const long long r0 = (long long)chunk * N / NCH;
  const long long r1 = (long long)(chunk + 1) * N / NCH;
  float s = 0.f, s2 = 0.f;
  for (long long n = r0; n < r1; ++n) {
    const float v = x[n * FP + c];
    s += v; s2 += v * v;
  }
  part[(size_t)chunk * (2 * FP) + c]      = s;
  part[(size_t)chunk * (2 * FP) + FP + c] = s2;
}

__global__ __launch_bounds__(192)
void k_bn_final(const float* __restrict__ part, int N,
                const float* __restrict__ g, const float* __restrict__ b,
                float* __restrict__ ss) {
  const int c = threadIdx.x;
  float s = 0.f, s2 = 0.f;
  for (int ch = 0; ch < NCH; ++ch) {
    s  += part[(size_t)ch * (2 * FP) + c];
    s2 += part[(size_t)ch * (2 * FP) + FP + c];
  }
  const float m   = s / (float)N;
  const float var = s2 / (float)N - m * m;
  float sc = 0.f, sh = 0.f;
  if (c < F) { sc = g[c] * rsqrtf(var + 1e-5f); sh = b[c] - m * sc; }
  ss[c] = sc; ss[FP + c] = sh;
}

// ---------------------------------------------------------------------------
// Epilogues
// ---------------------------------------------------------------------------
// feat = gemm + bias (+ lateral[lidx[n]]); write f32 [N,168] and bf16 [N,192]
__global__ void k_ep_feat(const float* __restrict__ gm, const float* __restrict__ bias,
                          const float* __restrict__ lat, const int* __restrict__ lidx,
                          float* __restrict__ featf, bf16* __restrict__ featb,
                          long long total) {
  long long i = (long long)blockIdx.x * blockDim.x + threadIdx.x;
  if (i >= total) return;
  const int c = (int)(i % FP);
  const long long n = i / FP;
  if (c < F) {
    float v = gm[n * FP + c] + bias[c];
    if (lat) v += lat[(size_t)lidx[n] * F + c];
    featf[n * F + c] = v;
    featb[i] = (bf16)v;
  } else {
    featb[i] = (bf16)0.0f;
  }
}

// out = relu(cv*scale + shift), f32 [N,168]
__global__ void k_ep_bnrelu(const float* __restrict__ cv, const float* __restrict__ ss,
                            float* __restrict__ out, long long total) {
  long long i = (long long)blockIdx.x * blockDim.x + threadIdx.x;
  if (i >= total) return;
  const int c = (int)(i % F);
  const long long n = i / F;
  const float v = cv[n * FP + c] * ss[c] + ss[FP + c];
  out[i] = v > 0.f ? v : 0.f;
}

// out7 = relu(bn(cv7)) + out6[p7[n]] + out5[p6[p7[n]]]
__global__ void k_ep_fuse7(const float* __restrict__ cv7, const float* __restrict__ ss,
                           const float* __restrict__ out6, const float* __restrict__ out5,
                           const int* __restrict__ p7, const int* __restrict__ p6,
                           float* __restrict__ out7, long long total) {
  long long i = (long long)blockIdx.x * blockDim.x + threadIdx.x;
  if (i >= total) return;
  const int c = (int)(i % F);
  const long long n = i / F;
  float v = cv7[n * FP + c] * ss[c] + ss[FP + c];
  v = v > 0.f ? v : 0.f;
  const int j6 = p7[n];
  const int j5 = p6[j6];
  out7[i] = v + out6[(size_t)j6 * F + c] + out5[(size_t)j5 * F + c];
}

// dst[n,:] = bf16(src[idx[n],:]) with zero pad  (src f32 ld 168)
__global__ void k_gather_pad(const float* __restrict__ src, const int* __restrict__ idx,
                             bf16* __restrict__ dst, long long total) {
  long long i = (long long)blockIdx.x * blockDim.x + threadIdx.x;
  if (i >= total) return;
  const int c = (int)(i % FP);
  const long long n = i / FP;
  dst[i] = (c < F) ? (bf16)src[(size_t)idx[n] * F + c] : (bf16)0.0f;
}

// dst[n,:] = bf16(relu(cv[idx?idx[n]:n, :]*scale + shift)) with zero pad (cv f32 ld 192)
__global__ void k_ep_bnrelu_gather_bf16(const float* __restrict__ cv,
                                        const float* __restrict__ ss,
                                        const int* __restrict__ idx,
                                        bf16* __restrict__ dst, long long total) {
  long long i = (long long)blockIdx.x * blockDim.x + threadIdx.x;
  if (i >= total) return;
  const int c = (int)(i % FP);
  const long long n = i / FP;
  if (c < F) {
    const long long r = idx ? (long long)idx[n] : n;
    float v = cv[r * FP + c] * ss[c] + ss[FP + c];
    dst[i] = (bf16)(v > 0.f ? v : 0.f);
  } else {
    dst[i] = (bf16)0.0f;
  }
}

// d_out[n,c] = g2[n, c] + cb2[c]   (g2 f32 ld 32, c < 21)
__global__ void k_ep_final(const float* __restrict__ g2, const float* __restrict__ cb2,
                           float* __restrict__ out, long long total) {
  long long i = (long long)blockIdx.x * blockDim.x + threadIdx.x;
  if (i >= total) return;
  const int c = (int)(i % OUTC);
  const long long n = i / OUTC;
  out[i] = g2[n * OUTP + c] + cb2[c];
}

// ---------------------------------------------------------------------------
// Host launcher
// ---------------------------------------------------------------------------
extern "C" void kernel_launch(void* const* d_in, const int* in_sizes, int n_in,
                              void* d_out, int out_size, void* d_ws, size_t ws_size,
                              hipStream_t stream) {
  (void)in_sizes; (void)n_in; (void)out_size; (void)ws_size;

  const float* features5 = (const float*)d_in[0];
  const float* features6 = (const float*)d_in[1];
  const float* features7 = (const float*)d_in[2];
  const int*   pidx6     = (const int*)d_in[3];
  const int*   pidx7     = (const int*)d_in[4];
  const int*   pidx8     = (const int*)d_in[5];
  const int*   neigh5    = (const int*)d_in[6];
  const int*   neigh6    = (const int*)d_in[7];
  const int*   neigh7    = (const int*)d_in[8];
  const int*   neigh8    = (const int*)d_in[9];
  const int*   qidx      = (const int*)d_in[10];
  const float* w0 = (const float*)d_in[11]; const float* b0 = (const float*)d_in[12];
  const float* w1 = (const float*)d_in[13]; const float* b1 = (const float*)d_in[14];
  const float* w2 = (const float*)d_in[15]; const float* b2 = (const float*)d_in[16];
  const float* Wc0 = (const float*)d_in[17]; const float* gc0 = (const float*)d_in[18]; const float* bc0 = (const float*)d_in[19];
  const float* Wc1 = (const float*)d_in[20]; const float* gc1 = (const float*)d_in[21]; const float* bc1 = (const float*)d_in[22];
  const float* Wc2 = (const float*)d_in[23]; const float* gc2 = (const float*)d_in[24]; const float* bc2 = (const float*)d_in[25];
  const float* Wup = (const float*)d_in[26]; const float* gup = (const float*)d_in[27]; const float* bup = (const float*)d_in[28];
  const float* cw1 = (const float*)d_in[29]; /* cb1 (zeros; cancels through batch-stat BN) */
  const float* cg  = (const float*)d_in[31]; const float* cbeta = (const float*)d_in[32];
  const float* cw2 = (const float*)d_in[33]; const float* cb2 = (const float*)d_in[34];
  float* out_final = (float*)d_out;

  // ---- bump allocator over d_ws ----
  char* base = (char*)d_ws;
  size_t off = 0;
  auto alloc = [&](size_t bytes) -> void* {
    void* p = base + off;
    off = (off + bytes + 255) & ~size_t(255);
    return p;
  };

  // swizzled weights: taps * ck * NT * 512 bf16 elements
  bf16* WT0  = (bf16*)alloc((size_t)1 * 8 * 12 * 512 * 2);
  bf16* WT1  = (bf16*)alloc((size_t)1 * 4 * 12 * 512 * 2);
  bf16* WT2  = (bf16*)alloc((size_t)1 * 2 * 12 * 512 * 2);
  bf16* WTC0 = (bf16*)alloc((size_t)KT * 6 * 12 * 512 * 2);
  bf16* WTC1 = (bf16*)alloc((size_t)KT * 6 * 12 * 512 * 2);
  bf16* WTC2 = (bf16*)alloc((size_t)KT * 6 * 12 * 512 * 2);
  bf16* WTUP = (bf16*)alloc((size_t)KT * 6 * 12 * 512 * 2);
  bf16* WCL1 = (bf16*)alloc((size_t)1 * 6 * 12 * 512 * 2);
  bf16* WCL2 = (bf16*)alloc((size_t)1 * 6 * 2 * 512 * 2);
  float* part = (float*)alloc((size_t)NCH * 2 * FP * 4);
  float* ss0 = (float*)alloc(2 * FP * 4);
  float* ss1 = (float*)alloc(2 * FP * 4);
  float* ss2 = (float*)alloc(2 * FP * 4);
  float* ss3 = (float*)alloc(2 * FP * 4);
  float* ss4 = (float*)alloc(2 * FP * 4);

  bf16*  xb5in = (bf16*)alloc((size_t)N5 * C5 * 2);
  float* cv5   = (float*)alloc((size_t)N5 * FP * 4);
  float* feat5 = (float*)alloc((size_t)N5 * F * 4);
  bf16*  xb5   = (bf16*)alloc((size_t)N5 * FP * 2);
  float* out5  = (float*)alloc((size_t)N5 * F * 4);

  bf16*  xb6in = (bf16*)alloc((size_t)N6 * C6 * 2);
  float* cv6   = (float*)alloc((size_t)N6 * FP * 4);
  float* feat6 = (float*)alloc((size_t)N6 * F * 4);
  bf16*  xb6   = (bf16*)alloc((size_t)N6 * FP * 2);
  float* out6  = (float*)alloc((size_t)N6 * F * 4);

  bf16*  xb7in = (bf16*)alloc((size_t)N7 * C7 * 2);
  float* cv7   = (float*)alloc((size_t)N7 * FP * 4);
  float* feat7 = (float*)alloc((size_t)N7 * F * 4);
  bf16*  xb7   = (bf16*)alloc((size_t)N7 * FP * 2);
  float* out7  = (float*)alloc((size_t)N7 * F * 4);

  bf16*  xb8 = (bf16*)alloc((size_t)N8 * FP * 2);
  float* cv8 = (float*)alloc((size_t)N8 * FP * 4);   // reused as classifier GEMM1 out
  bf16*  xq  = (bf16*)alloc((size_t)Q * FP * 2);
  bf16*  xh  = (bf16*)alloc((size_t)Q * FP * 2);
  float* g2  = (float*)alloc((size_t)Q * OUTP * 4);
  float* g1  = cv8;  // cv8 dead after xq is built

  auto grid1 = [](long long total) { return dim3((unsigned)((total + 255) / 256)); };
  auto gridW = [](int n) { return dim3((unsigned)((n + 127) / 128)); };  // 8 waves * 16 rows
  auto shm   = [](int ntaps, int ck, int nt) {
    return (size_t)(ntaps > 1 ? 2 : 1) * (size_t)ck * nt * 512 * 2;
  };

  // ---- weight repack (fragment swizzle) + input conversion ----
  { long long t = 1LL*8*12*512;    k_repack_sw<<<grid1(t),256,0,stream>>>(w0,  WT0,  t, C5, F, 8, 12); }
  { long long t = 1LL*4*12*512;    k_repack_sw<<<grid1(t),256,0,stream>>>(w1,  WT1,  t, C6, F, 4, 12); }
  { long long t = 1LL*2*12*512;    k_repack_sw<<<grid1(t),256,0,stream>>>(w2,  WT2,  t, C7, F, 2, 12); }
  { long long t = (long long)KT*6*12*512; k_repack_sw<<<grid1(t),256,0,stream>>>(Wc0, WTC0, t, F, F, 6, 12); }
  { long long t = (long long)KT*6*12*512; k_repack_sw<<<grid1(t),256,0,stream>>>(Wc1, WTC1, t, F, F, 6, 12); }
  { long long t = (long long)KT*6*12*512; k_repack_sw<<<grid1(t),256,0,stream>>>(Wc2, WTC2, t, F, F, 6, 12); }
  { long long t = (long long)KT*6*12*512; k_repack_sw<<<grid1(t),256,0,stream>>>(Wup, WTUP, t, F, F, 6, 12); }
  { long long t = 1LL*6*12*512;    k_repack_sw<<<grid1(t),256,0,stream>>>(cw1, WCL1, t, F, F,    6, 12); }
  { long long t = 1LL*6*2*512;     k_repack_sw<<<grid1(t),256,0,stream>>>(cw2, WCL2, t, F, OUTC, 6, 2); }
  { long long t = (long long)N5*C5; k_cvt_pad<<<grid1(t),256,0,stream>>>(features5, xb5in, t, C5, C5); }
  { long long t = (long long)N6*C6; k_cvt_pad<<<grid1(t),256,0,stream>>>(features6, xb6in, t, C6, C6); }
  { long long t = (long long)N7*C7; k_cvt_pad<<<grid1(t),256,0,stream>>>(features7, xb7in, t, C7, C7); }

  auto bn = [&](const float* cv, int N, const float* g, const float* b, float* ss) {
    k_bn_part<<<dim3(NCH), 192, 0, stream>>>(cv, N, part);
    k_bn_final<<<dim3(1), 192, 0, stream>>>(part, N, g, b, ss);
  };

  // ---- depth 5 ----
  k_gconv<12><<<gridW(N5),256,shm(1,8,12),stream>>>(xb5in, nullptr, 1, WT0, cv5, N5, 8);
  k_ep_feat<<<grid1((long long)N5*FP),256,0,stream>>>(cv5, b0, nullptr, nullptr, feat5, xb5, (long long)N5*FP);
  k_gconv<12><<<gridW(N5),256,shm(KT,6,12),stream>>>(xb5, neigh5, KT, WTC0, cv5, N5, 6);
  bn(cv5, N5, gc0, bc0, ss0);
  k_ep_bnrelu<<<grid1((long long)N5*F),256,0,stream>>>(cv5, ss0, out5, (long long)N5*F);

  // ---- depth 6 ----
  k_gconv<12><<<gridW(N6),256,shm(1,4,12),stream>>>(xb6in, nullptr, 1, WT1, cv6, N6, 4);
  k_ep_feat<<<grid1((long long)N6*FP),256,0,stream>>>(cv6, b1, feat5, pidx6, feat6, xb6, (long long)N6*FP);
  k_gconv<12><<<gridW(N6),256,shm(KT,6,12),stream>>>(xb6, neigh6, KT, WTC1, cv6, N6, 6);
  bn(cv6, N6, gc1, bc1, ss1);
  k_ep_bnrelu<<<grid1((long long)N6*F),256,0,stream>>>(cv6, ss1, out6, (long long)N6*F);

  // ---- depth 7 ----
  k_gconv<12><<<gridW(N7),256,shm(1,2,12),stream>>>(xb7in, nullptr, 1, WT2, cv7, N7, 2);
  k_ep_feat<<<grid1((long long)N7*FP),256,0,stream>>>(cv7, b2, feat6, pidx7, feat7, xb7, (long long)N7*FP);
  k_gconv<12><<<gridW(N7),256,shm(KT,6,12),stream>>>(xb7, neigh7, KT, WTC2, cv7, N7, 6);
  bn(cv7, N7, gc2, bc2, ss2);
  k_ep_fuse7<<<grid1((long long)N7*F),256,0,stream>>>(cv7, ss2, out6, out5, pidx7, pidx6, out7, (long long)N7*F);

  // ---- depth 8 (deconv) ----
  k_gather_pad<<<grid1((long long)N8*FP),256,0,stream>>>(out7, pidx8, xb8, (long long)N8*FP);
  k_gconv<12><<<gridW(N8),256,shm(KT,6,12),stream>>>(xb8, neigh8, KT, WTUP, cv8, N8, 6);
  bn(cv8, N8, gup, bup, ss3);
  k_ep_bnrelu_gather_bf16<<<grid1((long long)Q*FP),256,0,stream>>>(cv8, ss3, qidx, xq, (long long)Q*FP);

  // ---- classifier ----
  k_gconv<12><<<gridW(Q),256,shm(1,6,12),stream>>>(xq, nullptr, 1, WCL1, g1, Q, 6);
  bn(g1, Q, cg, cbeta, ss4);
  k_ep_bnrelu_gather_bf16<<<grid1((long long)Q*FP),256,0,stream>>>(g1, ss4, nullptr, xh, (long long)Q*FP);
  k_gconv<2><<<gridW(Q),256,shm(1,6,2),stream>>>(xh, nullptr, 1, WCL2, g2, Q, 6);
  k_ep_final<<<grid1((long long)Q*OUTC),256,0,stream>>>(g2, cb2, out_final, (long long)Q*OUTC);
}
